// IndependentQuantileHead_30657476559311
// MI455X (gfx1250) — matrix operand order repaired and checked by reading, not compiled, not run on hardware
//
#include <hip/hip_runtime.h>
#include <hip/hip_bf16.h>

// ---------------- problem constants ----------------
#define HIDDEN   2048
#define HALF     1024
#define QUANT    16
#define BATCH    8192

// ---------------- tiling ----------------
#define BM 256          // batch rows per workgroup
#define BN 128          // qk columns per workgroup (inside one head: 1024 % 128 == 0)
#define KB 32           // k-step (bf16 WMMA K)
#define NKT (HIDDEN / KB)
#define NTHREADS 512    // 16 waves (wave32), 4x4 wave grid, 64x32 C patch per wave

typedef __bf16 bf16_t;
typedef bf16_t v16bf __attribute__((ext_vector_type(16)));
typedef float  v8f   __attribute__((ext_vector_type(8)));

union FragU { v16bf v; uint4 q[2]; };

__device__ __forceinline__ void cvt_split(float f, unsigned short& hi, unsigned short& lo) {
    bf16_t h = (bf16_t)f;                 // RNE f32 -> bf16
    float  fh = (float)h;
    bf16_t l = (bf16_t)(f - fh);          // residual
    hi = __builtin_bit_cast(unsigned short, h);
    lo = __builtin_bit_cast(unsigned short, l);
}

__global__ void __launch_bounds__(256) iqh_init_out(const float* __restrict__ b2,
                                                    float* __restrict__ out) {
    int i = blockIdx.x * blockDim.x + threadIdx.x;
    if (i < BATCH * QUANT) out[i] = b2[i & (QUANT - 1)];
}

// grid: (16384/BN = 128, BATCH/BM = 32), block: 512 threads = 16 waves
__global__ void __launch_bounds__(NTHREADS)
iqh_gemm_gelu_kernel(const float* __restrict__ x,
                     const float* __restrict__ W1,
                     const float* __restrict__ b1,
                     const float* __restrict__ W2,
                     float* __restrict__ out) {
    // ---- LDS: double-buffered, fragment-packed bf16 tiles (hi part, then lo part) ----
    // A buffer: [2 parts][16 m-tiles][32 lanes][16 bf16] = 32 KB  -> x2 buffers = 64 KB
    // B buffer: [2 parts][ 8 n-tiles][32 lanes][16 bf16] = 16 KB  -> x2 buffers = 32 KB
    const int A_PART = 16 * 32 * 16;      // 8192 ushorts
    const int A_BUF  = 2 * A_PART;        // 16384 ushorts
    const int B_PART = 8 * 32 * 16;       // 4096 ushorts
    const int B_BUF  = 2 * B_PART;        // 8192 ushorts
    __shared__ __align__(16) unsigned short ldsA[2 * A_BUF];
    __shared__ __align__(16) unsigned short ldsB[2 * B_BUF];

    const int tid = threadIdx.x;
    const int m0  = blockIdx.y * BM;                  // batch row base
    const int q   = blockIdx.x >> 3;                  // quantile head (128 n-tiles / 8 per head)
    const int nb  = (blockIdx.x & 7) * BN;            // column base inside head
    const float* __restrict__ W1q = W1 + (size_t)q * HIDDEN * HALF;

    // ---- per-wave identities: 4x4 wave grid, each wave = 64 rows x 32 cols ----
    const int wv   = tid >> 5;            // 0..15
    const int L    = tid & 31;            // lane
    const int wrow = wv >> 2;             // 0..3  (64 rows each)
    const int wcol = wv & 3;              // 0..3  (32 cols each)

    // ---- global-load register staging ----
    float4 xr[4];                         // x tile: 256 rows x 32 k, 4 passes of 64 rows
    float4 wr4[2];                        // W1 tile: 32 k x 128 n, 2 passes of 16 k-rows

    const int xrow = tid >> 3;            // 0..63 (row within pass)
    const int xk0  = (tid & 7) * 4;       // k offset 0..28
    const int wkr  = tid >> 5;            // 0..15 (k-row within pass)
    const int wn0  = (tid & 31) * 4;      // n offset 0..124

    auto loadX = [&](int kt) {
#pragma unroll
        for (int p = 0; p < 4; ++p)
            xr[p] = *(const float4*)(x + (size_t)(m0 + p * 64 + xrow) * HIDDEN + kt * KB + xk0);
    };
    auto loadW = [&](int kt) {
#pragma unroll
        for (int p = 0; p < 2; ++p)
            wr4[p] = *(const float4*)(W1q + (size_t)(kt * KB + p * 16 + wkr) * HALF + nb + wn0);
    };

    auto storeX = [&](int buf) {
        // A fragment layout (16-bit A 16x32, matching B):
        //   lane-half = k/16 ; idx = k%16
        const int lane = (xrow & 15) + 16 * (xk0 >> 4);
        const int idx0 = xk0 & 15;
        const int base = buf * A_BUF;
#pragma unroll
        for (int p = 0; p < 4; ++p) {
            const int lr = p * 64 + xrow;
            const int mt = lr >> 4;
            float v[4] = {xr[p].x, xr[p].y, xr[p].z, xr[p].w};
            unsigned short h[4], l[4];
#pragma unroll
            for (int e = 0; e < 4; ++e) cvt_split(v[e], h[e], l[e]);
            uint2 ph, pl;
            ph.x = (unsigned)h[0] | ((unsigned)h[1] << 16);
            ph.y = (unsigned)h[2] | ((unsigned)h[3] << 16);
            pl.x = (unsigned)l[0] | ((unsigned)l[1] << 16);
            pl.y = (unsigned)l[2] | ((unsigned)l[3] << 16);
            const int off = base + (mt * 32 + lane) * 16 + idx0;  // idx0 % 4 == 0 -> 8B aligned
            *(uint2*)&ldsA[off]          = ph;
            *(uint2*)&ldsA[A_PART + off] = pl;
        }
    };
    auto storeW = [&](int buf) {
        // B fragment layout (16-bit B 32x16): lane = (n%16) + 16*(k/16), idx = k%16
        const int nt   = wn0 >> 4;
        const int base = buf * B_BUF;
#pragma unroll
        for (int p = 0; p < 2; ++p) {
            const int kr    = p * 16 + wkr;
            const int lane0 = (wn0 & 15) + 16 * (kr >> 4);
            const int idx   = kr & 15;
            float v[4] = {wr4[p].x, wr4[p].y, wr4[p].z, wr4[p].w};
#pragma unroll
            for (int e = 0; e < 4; ++e) {
                unsigned short h, l;
                cvt_split(v[e], h, l);
                const int off = base + (nt * 32 + lane0 + e) * 16 + idx;
                ldsB[off]          = h;
                ldsB[B_PART + off] = l;
            }
        }
    };

    // ---- accumulators: 4 m-tiles x 2 n-tiles of 16x16 f32 ----
    v8f c[4][2];
#pragma unroll
    for (int mi = 0; mi < 4; ++mi)
#pragma unroll
        for (int ni = 0; ni < 2; ++ni) c[mi][ni] = (v8f)0.0f;

    // prologue: tile 0 -> regs -> LDS buffer 0
    loadX(0);
    loadW(0);
    storeX(0);
    storeW(0);
    __syncthreads();

    for (int kt = 0; kt < NKT; ++kt) {
        const int cur  = kt & 1;
        const int nxt  = cur ^ 1;
        const bool more = (kt + 1 < NKT);

        if (more) {                       // issue next tile's global loads early
            loadX(kt + 1);
            loadW(kt + 1);
        }

        // A fragments resident across both n-tiles
        FragU ah[4], al[4];
#pragma unroll
        for (int mi = 0; mi < 4; ++mi) {
            const int base = cur * A_BUF + ((wrow * 4 + mi) * 32 + L) * 16;
            ah[mi].q[0] = *(const uint4*)&ldsA[base];
            ah[mi].q[1] = *(const uint4*)&ldsA[base + 8];
            al[mi].q[0] = *(const uint4*)&ldsA[A_PART + base];
            al[mi].q[1] = *(const uint4*)&ldsA[A_PART + base + 8];
        }
#pragma unroll
        for (int ni = 0; ni < 2; ++ni) {
            const int base = cur * B_BUF + ((wcol * 2 + ni) * 32 + L) * 16;
            FragU bh, bl;
            bh.q[0] = *(const uint4*)&ldsB[base];
            bh.q[1] = *(const uint4*)&ldsB[base + 8];
            bl.q[0] = *(const uint4*)&ldsB[B_PART + base];
            bl.q[1] = *(const uint4*)&ldsB[B_PART + base + 8];
#pragma unroll
            for (int mi = 0; mi < 4; ++mi) {
                // bf16x3: hi*hi + hi*lo + lo*hi accumulated in f32
                c[mi][ni] = __builtin_amdgcn_wmma_f32_16x16x32_bf16(
                    false, ah[mi].v, false, bh.v, (short)0, c[mi][ni], false, false);
                c[mi][ni] = __builtin_amdgcn_wmma_f32_16x16x32_bf16(
                    false, ah[mi].v, false, bl.v, (short)0, c[mi][ni], false, false);
                c[mi][ni] = __builtin_amdgcn_wmma_f32_16x16x32_bf16(
                    false, al[mi].v, false, bh.v, (short)0, c[mi][ni], false, false);
            }
        }

        if (more) {                       // convert + store next tile into other buffer
            storeX(nxt);
            storeW(nxt);
        }
        __syncthreads();                  // single barrier per k-step (double buffered)
    }

    // ---- fused epilogue: +b1, exact GELU, *W2, reduce over N, atomic into out ----
    float b1v[2], w2v[2];
#pragma unroll
    for (int ni = 0; ni < 2; ++ni) {
        const int kk = nb + wcol * 32 + ni * 16 + (L & 15);   // column inside head
        b1v[ni] = b1[q * HALF + kk];
        w2v[ni] = W2[q * HALF + kk];
    }

    const int half = L >> 4;   // C layout: VGPR i -> M = i + 8*half, N = L%16
#pragma unroll
    for (int mi = 0; mi < 4; ++mi) {
#pragma unroll
        for (int i = 0; i < 8; ++i) {
            float acc = 0.0f;
#pragma unroll
            for (int ni = 0; ni < 2; ++ni) {
                float hv = c[mi][ni][i] + b1v[ni];
                float g  = 0.5f * hv * (1.0f + erff(hv * 0.70710678118654752f)); // exact GELU
                acc += g * w2v[ni];
            }
            // reduce across the 16 lanes sharing this row (xor stays within 16-lane halves)
#pragma unroll
            for (int o = 8; o >= 1; o >>= 1) acc += __shfl_xor(acc, o, 32);
            if ((L & 15) == 0) {
                const int row = m0 + wrow * 64 + mi * 16 + i + 8 * half;
                atomicAdd(&out[row * QUANT + q], acc);
            }
        }
    }
}

extern "C" void kernel_launch(void* const* d_in, const int* in_sizes, int n_in,
                              void* d_out, int out_size, void* d_ws, size_t ws_size,
                              hipStream_t stream) {
    const float* x  = (const float*)d_in[0];
    const float* W1 = (const float*)d_in[1];
    const float* b1 = (const float*)d_in[2];
    const float* W2 = (const float*)d_in[3];
    const float* b2 = (const float*)d_in[4];
    float* out = (float*)d_out;

    iqh_init_out<<<(BATCH * QUANT + 255) / 256, 256, 0, stream>>>(b2, out);

    dim3 grid((QUANT * HALF) / BN, BATCH / BM);   // (128, 32)
    iqh_gemm_gelu_kernel<<<grid, NTHREADS, 0, stream>>>(x, W1, b1, W2, out);
}
